// LearningProgressIM_55868934586377
// MI455X (gfx1250) — matrix operand order repaired and checked by reading, not compiled, not run on hardware
//
#include <hip/hip_runtime.h>

// Problem constants from the reference (N=65536, D=256, B=2048).
#define NTOT  65536
#define DIM   256
#define BATCH 2048
#define NPREV (NTOT - BATCH)   // 63488
#define MT    4                // M-tiles (16 targets each) per block in bf16 path
#define MROWS (MT * 16)        // 64 targets per block
#define NT    2                // N-strips (16 lib rows each) per wave
#define NROWS (8 * 16 * NT)    // 256 library rows per block

typedef __attribute__((ext_vector_type(16))) __bf16 v16bf;
typedef __attribute__((ext_vector_type(8)))  float  v8f;

union V16U {
    v16bf          v;
    unsigned short u[16];
    uint4          q[2];
};

// ---- workspace layout (fast path) ----
#define WS_SLOTS_BYTES ((size_t)(2 * BATCH) * 8)                 // 32 KB
#define WS_NORMS_OFF   ((size_t)32 * 1024)
#define WS_NORMS_BYTES ((size_t)NTOT * 4)                        // 256 KB
#define WS_TGT_HI_OFF  (WS_NORMS_OFF + WS_NORMS_BYTES)
#define WS_TGT_BYTES   ((size_t)BATCH * DIM * 2)                 // 1 MB
#define WS_TGT_LO_OFF  (WS_TGT_HI_OFF + WS_TGT_BYTES)
#define WS_LIB_HI_OFF  (WS_TGT_LO_OFF + WS_TGT_BYTES)
#define WS_LIB_BYTES   ((size_t)NTOT * DIM * 2)                  // 32 MB
#define WS_LIB_LO_OFF  (WS_LIB_HI_OFF + WS_LIB_BYTES)
#define WS_TOTAL       (WS_LIB_LO_OFF + WS_LIB_BYTES)            // ~66.3 MB

__device__ __forceinline__ unsigned short bf16_hi_bits(float x) {
    return (unsigned short)(__float_as_uint(x) >> 16);           // truncate
}
__device__ __forceinline__ float bf16_hi_float(float x) {
    return __uint_as_float(__float_as_uint(x) & 0xFFFF0000u);
}
__device__ __forceinline__ unsigned short bf16_rne_bits(float x) {
    unsigned u = __float_as_uint(x);
    u += 0x7FFFu + ((u >> 16) & 1u);                             // round-nearest-even
    return (unsigned short)(u >> 16);
}

__device__ __forceinline__ unsigned long long shflxor_u64(unsigned long long v, int m) {
    unsigned lo = (unsigned)v;
    unsigned hi = (unsigned)(v >> 32);
    lo = __shfl_xor(lo, m, 32);
    hi = __shfl_xor(hi, m, 32);
    return ((unsigned long long)hi << 32) | (unsigned long long)lo;
}

// XOR-swizzled LDS offset (16-byte granules): conflict-free 16-lane b128 reads
// without row padding. `s` must be granule (8-short) aligned.
__device__ __forceinline__ int aswz(int row, int s) {
    int g = (s >> 3) ^ (row & 15);
    return row * 256 + (g << 3);
}

// ---------------------------------------------------------------------------
// Init: workspace slots (packed (key<<32)|idx) to +inf.
// ---------------------------------------------------------------------------
__global__ void lp_init_slots(unsigned long long* __restrict__ slots) {
    int i = blockIdx.x * blockDim.x + threadIdx.x;
    if (i < 2 * BATCH) slots[i] = ~0ull;
}

// ---------------------------------------------------------------------------
// One-shot fp32 -> bf16 hi/lo split (hi = trunc so the residual is exact,
// lo = RNE of the residual). 4 elements per thread.
// ---------------------------------------------------------------------------
__global__ __launch_bounds__(256)
void lp_convert(const float* __restrict__ src,
                unsigned short* __restrict__ hi,
                unsigned short* __restrict__ lo,
                int n4)
{
    int i = blockIdx.x * blockDim.x + threadIdx.x;
    if (i >= n4) return;
    float4 x = ((const float4*)src)[i];

    unsigned h01 = (unsigned)bf16_hi_bits(x.x) | ((__float_as_uint(x.y)) & 0xFFFF0000u);
    unsigned h23 = (unsigned)bf16_hi_bits(x.z) | ((__float_as_uint(x.w)) & 0xFFFF0000u);

    float r0 = x.x - bf16_hi_float(x.x);
    float r1 = x.y - bf16_hi_float(x.y);
    float r2 = x.z - bf16_hi_float(x.z);
    float r3 = x.w - bf16_hi_float(x.w);
    unsigned l01 = (unsigned)bf16_rne_bits(r0) | ((unsigned)bf16_rne_bits(r1) << 16);
    unsigned l23 = (unsigned)bf16_rne_bits(r2) | ((unsigned)bf16_rne_bits(r3) << 16);

    uint2 hv; hv.x = h01; hv.y = h23;
    uint2 lv; lv.x = l01; lv.y = l23;
    ((uint2*)hi)[i] = hv;
    ((uint2*)lo)[i] = lv;
}

// ---------------------------------------------------------------------------
// Exact fp32 row norms of the library (one wave per row).
// ---------------------------------------------------------------------------
__global__ __launch_bounds__(256)
void lp_rownorm(const float* __restrict__ lib, float* __restrict__ norms)
{
    const int wave = threadIdx.x >> 5, lane = threadIdx.x & 31;
    const int row  = blockIdx.x * 8 + wave;
    const float4* p = (const float4*)(lib + (size_t)row * DIM + lane * 8);
    float4 a = p[0], b = p[1];
    float s = a.x*a.x + a.y*a.y + a.z*a.z + a.w*a.w
            + b.x*b.x + b.y*b.y + b.z*b.z + b.w*b.w;
    #pragma unroll
    for (int off = 16; off > 0; off >>= 1) s += __shfl_xor(s, off, 32);
    if (lane == 0) norms[row] = s;
}

// ---------------------------------------------------------------------------
// FAST PATH: pure bf16 WMMA NN search from preconverted data.
// Block: 64 targets (4 M-tiles in swizzled LDS) x 256 library rows
// (8 waves x 2 N-strips). Each A fragment read from LDS feeds 6 WMMAs;
// each pair of B strips is 8 global b128 loads per k-chunk feeding 24 WMMAs.
// Score = ||lib||^2 - 2*dot (||t||^2 constant per row); argmin via packed
// (monotone-float<<32 | idx) u64 atomic min -> ties pick smallest index.
// ---------------------------------------------------------------------------
__global__ __launch_bounds__(256)
void lp_nn_bf16(const unsigned short* __restrict__ tgtHi,
                const unsigned short* __restrict__ tgtLo,
                const unsigned short* __restrict__ libHi,
                const unsigned short* __restrict__ libLo,
                const float* __restrict__ norms,
                unsigned libBase,
                unsigned long long* __restrict__ slots)
{
    __shared__ unsigned short AhiL[MROWS * 256];   // 32 KB
    __shared__ unsigned short AloL[MROWS * 256];   // 32 KB  (64 KB total)

    const int t     = threadIdx.x;
    const int mBase = blockIdx.y * MROWS;

    // ---- Stage A tile (already bf16) into swizzled LDS ----
    {
        const int row = t >> 2;          // 64 rows, 4 threads each
        const int qb  = (t & 3) * 8;     // 8 granules (of 32) per thread
        const uint4* sh = (const uint4*)(tgtHi + (size_t)(mBase + row) * DIM);
        const uint4* sl = (const uint4*)(tgtLo + (size_t)(mBase + row) * DIM);
        #pragma unroll
        for (int j = 0; j < 8; ++j) {
            const int g  = qb + j;
            const int pg = g ^ (row & 15);
            *(uint4*)&AhiL[row * 256 + pg * 8] = sh[g];
            *(uint4*)&AloL[row * 256 + pg * 8] = sl[g];
        }
    }
    __syncthreads();

    const int wave = t >> 5;
    const int lane = t & 31;
    const int l    = lane & 15;
    const int h    = lane >> 4;

    const unsigned nIdx0   = blockIdx.x * (unsigned)NROWS + (unsigned)(wave * 32 + l);
    const unsigned nIdx1   = nIdx0 + 16;
    const unsigned rowAbs0 = libBase + nIdx0;
    const unsigned rowAbs1 = libBase + nIdx1;
    const unsigned short* bh0 = libHi + (size_t)rowAbs0 * DIM;
    const unsigned short* bl0 = libLo + (size_t)rowAbs0 * DIM;
    const unsigned short* bh1 = libHi + (size_t)rowAbs1 * DIM;
    const unsigned short* bl1 = libLo + (size_t)rowAbs1 * DIM;

    v8f zero = {};
    v8f acc0[MT], acc1[MT];
    #pragma unroll
    for (int m = 0; m < MT; ++m) { acc0[m] = zero; acc1[m] = zero; }

    for (int kc = 0; kc < 8; ++kc) {
        const int kb = kc * 32;

        // B fragments: lane holds 16 contiguous bf16 at K = kb + h*16
        V16U bhi0, blo0, bhi1, blo1;
        {
            const uint4* q;
            q = (const uint4*)(bh0 + kb + h * 16); bhi0.q[0] = q[0]; bhi0.q[1] = q[1];
            q = (const uint4*)(bl0 + kb + h * 16); blo0.q[0] = q[0]; blo0.q[1] = q[1];
            q = (const uint4*)(bh1 + kb + h * 16); bhi1.q[0] = q[0]; bhi1.q[1] = q[1];
            q = (const uint4*)(bl1 + kb + h * 16); blo1.q[0] = q[0]; blo1.q[1] = q[1];
        }

        #pragma unroll
        for (int m = 0; m < MT; ++m) {
            const int row = m * 16 + l;
            V16U ahi, alo;
            ahi.q[0] = *(const uint4*)&AhiL[aswz(row, kb + h * 16)];
            ahi.q[1] = *(const uint4*)&AhiL[aswz(row, kb + h * 16 + 8)];
            alo.q[0] = *(const uint4*)&AloL[aswz(row, kb + h * 16)];
            alo.q[1] = *(const uint4*)&AloL[aswz(row, kb + h * 16 + 8)];

            acc0[m] = __builtin_amdgcn_wmma_f32_16x16x32_bf16(false, ahi.v, false, bhi0.v,
                                                              (short)0, acc0[m], false, false);
            acc0[m] = __builtin_amdgcn_wmma_f32_16x16x32_bf16(false, ahi.v, false, blo0.v,
                                                              (short)0, acc0[m], false, false);
            acc0[m] = __builtin_amdgcn_wmma_f32_16x16x32_bf16(false, alo.v, false, bhi0.v,
                                                              (short)0, acc0[m], false, false);

            acc1[m] = __builtin_amdgcn_wmma_f32_16x16x32_bf16(false, ahi.v, false, bhi1.v,
                                                              (short)0, acc1[m], false, false);
            acc1[m] = __builtin_amdgcn_wmma_f32_16x16x32_bf16(false, ahi.v, false, blo1.v,
                                                              (short)0, acc1[m], false, false);
            acc1[m] = __builtin_amdgcn_wmma_f32_16x16x32_bf16(false, alo.v, false, bhi1.v,
                                                              (short)0, acc1[m], false, false);
        }
    }

    // ---- Argmin epilogue: C layout row = r + 8h, col = l ----
    const float ln0 = norms[rowAbs0];
    const float ln1 = norms[rowAbs1];

    #pragma unroll
    for (int s = 0; s < NT; ++s) {
        const float    ln     = s ? ln1 : ln0;
        const unsigned rowAbs = s ? rowAbs1 : rowAbs0;
        #pragma unroll
        for (int m = 0; m < MT; ++m) {
            #pragma unroll
            for (int r = 0; r < 8; ++r) {
                float    d  = ln - 2.0f * (s ? acc1[m][r] : acc0[m][r]);
                unsigned db = __float_as_uint(d);
                db = (db & 0x80000000u) ? ~db : (db | 0x80000000u);
                unsigned long long key =
                    ((unsigned long long)db << 32) | (unsigned long long)rowAbs;
                #pragma unroll
                for (int off = 1; off < 16; off <<= 1) {
                    unsigned long long o = shflxor_u64(key, off);
                    key = (o < key) ? o : key;
                }
                if (l == 0) atomicMin(&slots[mBase + m * 16 + r + 8 * h], key);
            }
        }
    }
}

// ---------------------------------------------------------------------------
// FALLBACK PATH (small ws): fused fp32->bf16 conversion + WMMA, 16x128/block.
// ---------------------------------------------------------------------------
__global__ __launch_bounds__(256)
void lp_nn_fused(const float* __restrict__ targets,
                 const float* __restrict__ lib,
                 unsigned libBase,
                 unsigned long long* __restrict__ slots)
{
    __shared__ unsigned short Ahi[16][264];
    __shared__ unsigned short Alo[16][264];

    const int t     = threadIdx.x;
    const int mBase = blockIdx.y * 16;

    {
        const int row = t >> 4;
        const int cb  = (t & 15) << 4;
        const float* tp = targets + (size_t)(mBase + row) * DIM + cb;
        #pragma unroll
        for (int i = 0; i < 16; i += 2) {
            float x0 = tp[i], x1 = tp[i + 1];
            unsigned short h0 = bf16_hi_bits(x0), h1 = bf16_hi_bits(x1);
            unsigned short l0 = bf16_rne_bits(x0 - bf16_hi_float(x0));
            unsigned short l1 = bf16_rne_bits(x1 - bf16_hi_float(x1));
            *(unsigned*)&Ahi[row][cb + i] = (unsigned)h0 | ((unsigned)h1 << 16);
            *(unsigned*)&Alo[row][cb + i] = (unsigned)l0 | ((unsigned)l1 << 16);
        }
    }
    __syncthreads();

    const int wave = t >> 5, lane = t & 31, l = lane & 15, h = lane >> 4;
    const unsigned nIdx   = blockIdx.x * 128u + (unsigned)(wave * 16 + l);
    const unsigned rowAbs = libBase + nIdx;
    const float*   bp     = lib + (size_t)rowAbs * DIM;

    v8f   acc    = {};
    float lnPart = 0.0f;

    #pragma unroll
    for (int kc = 0; kc < 8; ++kc) {
        const int kb = kc * 32;
        float bv[16];
        {
            const float4* q = (const float4*)(bp + kb + h * 16);
            float4 q0 = q[0], q1 = q[1], q2 = q[2], q3 = q[3];
            bv[0]=q0.x; bv[1]=q0.y; bv[2]=q0.z;  bv[3]=q0.w;
            bv[4]=q1.x; bv[5]=q1.y; bv[6]=q1.z;  bv[7]=q1.w;
            bv[8]=q2.x; bv[9]=q2.y; bv[10]=q2.z; bv[11]=q2.w;
            bv[12]=q3.x; bv[13]=q3.y; bv[14]=q3.z; bv[15]=q3.w;
        }
        V16U bhi, blo;
        #pragma unroll
        for (int i = 0; i < 16; ++i) {
            float x = bv[i];
            lnPart  += x * x;
            bhi.u[i] = bf16_hi_bits(x);
            blo.u[i] = bf16_hi_bits(x - bf16_hi_float(x));   // trunc lo: cheaper
        }
        V16U ahi, alo;
        ahi.q[0] = *(const uint4*)&Ahi[l][kb + h * 16];
        ahi.q[1] = *(const uint4*)&Ahi[l][kb + h * 16 + 8];
        alo.q[0] = *(const uint4*)&Alo[l][kb + h * 16];
        alo.q[1] = *(const uint4*)&Alo[l][kb + h * 16 + 8];

        acc = __builtin_amdgcn_wmma_f32_16x16x32_bf16(false, ahi.v, false, bhi.v,
                                                      (short)0, acc, false, false);
        acc = __builtin_amdgcn_wmma_f32_16x16x32_bf16(false, ahi.v, false, blo.v,
                                                      (short)0, acc, false, false);
        acc = __builtin_amdgcn_wmma_f32_16x16x32_bf16(false, alo.v, false, bhi.v,
                                                      (short)0, acc, false, false);
    }

    const float ln = lnPart + __shfl_xor(lnPart, 16, 32);

    #pragma unroll
    for (int r = 0; r < 8; ++r) {
        float    d  = ln - 2.0f * acc[r];
        unsigned db = __float_as_uint(d);
        db = (db & 0x80000000u) ? ~db : (db | 0x80000000u);
        unsigned long long key =
            ((unsigned long long)db << 32) | (unsigned long long)rowAbs;
        #pragma unroll
        for (int off = 1; off < 16; off <<= 1) {
            unsigned long long o = shflxor_u64(key, off);
            key = (o < key) ? o : key;
        }
        if (l == 0) atomicMin(&slots[mBase + r + 8 * h], key);
    }
}

// ---------------------------------------------------------------------------
// Finalize: exact fp32 re-score of selected neighbors + value/grad.
// ---------------------------------------------------------------------------
__global__ __launch_bounds__(256)
void lp_finalize(const float* __restrict__ targets,
                 const float* __restrict__ lib,
                 const unsigned long long* __restrict__ slotsPrev,
                 const unsigned long long* __restrict__ slotsNew,
                 float* __restrict__ vals,
                 float* __restrict__ grads)
{
    __shared__ float r1s[8], r2s[8];
    const int b = blockIdx.x;
    const int d = threadIdx.x;

    const unsigned i1 = (unsigned)(slotsPrev[b] & 0xFFFFFFFFull);
    const unsigned i2 = (unsigned)(slotsNew[b]  & 0xFFFFFFFFull);

    const float tv  = targets[(size_t)b * DIM + d];
    const float df1 = tv - lib[(size_t)i1 * DIM + d];
    const float df2 = tv - lib[(size_t)i2 * DIM + d];

    float s1 = df1 * df1, s2 = df2 * df2;
    #pragma unroll
    for (int off = 16; off > 0; off >>= 1) {
        s1 += __shfl_xor(s1, off, 32);
        s2 += __shfl_xor(s2, off, 32);
    }
    const int wv = d >> 5, ln = d & 31;
    if (ln == 0) { r1s[wv] = s1; r2s[wv] = s2; }
    __syncthreads();

    float t1 = 0.f, t2 = 0.f;
    #pragma unroll
    for (int i = 0; i < 8; ++i) { t1 += r1s[i]; t2 += r2s[i]; }

    const float d1 = sqrtf(t1), d2 = sqrtf(t2);
    if (d == 0) vals[b] = d1 - d2;
    const float g1 = (d1 > 0.f) ? (df1 / d1) : 0.f;
    const float g2 = (d2 > 0.f) ? (df2 / d2) : 0.f;
    grads[(size_t)b * DIM + d] = g1 - g2;
}

// ---------------------------------------------------------------------------
extern "C" void kernel_launch(void* const* d_in, const int* in_sizes, int n_in,
                              void* d_out, int out_size, void* d_ws, size_t ws_size,
                              hipStream_t stream)
{
    (void)in_sizes; (void)n_in; (void)out_size;

    const float* target_lib  = (const float*)d_in[0];
    const float* reached_lib = (const float*)d_in[1];
    const float* targets     = target_lib + (size_t)NPREV * DIM;

    char* ws = (char*)d_ws;
    unsigned long long* slots     = (unsigned long long*)ws;
    unsigned long long* slotsPrev = slots;
    unsigned long long* slotsNew  = slots + BATCH;

    float* vals  = (float*)d_out;
    float* grads = vals + BATCH;

    lp_init_slots<<<dim3((2 * BATCH + 255) / 256), dim3(256), 0, stream>>>(slots);

    if (ws_size >= WS_TOTAL) {
        // -------- fast path: preconvert once, then pure-WMMA search --------
        float*          norms = (float*)(ws + WS_NORMS_OFF);
        unsigned short* tgtHi = (unsigned short*)(ws + WS_TGT_HI_OFF);
        unsigned short* tgtLo = (unsigned short*)(ws + WS_TGT_LO_OFF);
        unsigned short* libHi = (unsigned short*)(ws + WS_LIB_HI_OFF);
        unsigned short* libLo = (unsigned short*)(ws + WS_LIB_LO_OFF);

        lp_convert<<<dim3((NTOT  * DIM / 4) / 256), dim3(256), 0, stream>>>(
            reached_lib, libHi, libLo, NTOT * DIM / 4);
        lp_convert<<<dim3((BATCH * DIM / 4) / 256), dim3(256), 0, stream>>>(
            targets, tgtHi, tgtLo, BATCH * DIM / 4);
        lp_rownorm<<<dim3(NTOT / 8), dim3(256), 0, stream>>>(reached_lib, norms);

        lp_nn_bf16<<<dim3(NPREV / NROWS, BATCH / MROWS), dim3(256), 0, stream>>>(
            tgtHi, tgtLo, libHi, libLo, norms, 0u, slotsPrev);
        lp_nn_bf16<<<dim3(BATCH / NROWS, BATCH / MROWS), dim3(256), 0, stream>>>(
            tgtHi, tgtLo, libHi, libLo, norms, (unsigned)NPREV, slotsNew);
    } else {
        // -------- fallback: fused conversion + WMMA --------
        lp_nn_fused<<<dim3(NPREV / 128, BATCH / 16), dim3(256), 0, stream>>>(
            targets, reached_lib, 0u, slotsPrev);
        lp_nn_fused<<<dim3(BATCH / 128, BATCH / 16), dim3(256), 0, stream>>>(
            targets, reached_lib, (unsigned)NPREV, slotsNew);
    }

    lp_finalize<<<dim3(BATCH), dim3(256), 0, stream>>>(
        targets, reached_lib, slotsPrev, slotsNew, vals, grads);
}
